// GemmaForCausalLM_46832323396483
// MI455X (gfx1250) — matrix operand order repaired and checked by reading, not compiled, not run on hardware
//
#include <hip/hip_runtime.h>
#include <hip/hip_bf16.h>

// ---------------------------------------------------------------- types ----
typedef __attribute__((ext_vector_type(16))) __bf16 v16bf;
typedef __attribute__((ext_vector_type(8)))  __bf16 v8bf;
typedef __attribute__((ext_vector_type(8)))  float  v8f;
typedef __attribute__((ext_vector_type(4)))  int    v4i;

#define S_LEN 2048
#define HID   2304
#define NHEAD 8
#define NKV   4
#define HD    256
#define FFDIM 9216
#define QKV_N 4096
#define WIN   1024
#define NEGF  (-1.0e30f)

// ----------------------------------------------- CDNA5 async global->LDS ---
#if defined(__has_builtin)
#  if __has_builtin(__builtin_amdgcn_global_load_async_to_lds_b128)
#    define HAVE_ASYNC_LDS 1
#  endif
#  if __has_builtin(__builtin_amdgcn_s_wait_asynccnt)
#    define HAVE_WAIT_ASYNC 1
#  endif
#endif

#ifdef HAVE_ASYNC_LDS
typedef __attribute__((address_space(1))) v4i as1_v4i;   // global (AS1)
typedef __attribute__((address_space(3))) v4i as3_v4i;   // LDS (AS3)
#endif

__device__ __forceinline__ void stage16(const unsigned short* src,
                                        unsigned short* lds) {
#ifdef HAVE_ASYNC_LDS
    __builtin_amdgcn_global_load_async_to_lds_b128((as1_v4i*)src, (as3_v4i*)lds,
                                                   0, 0);
#else
    *(v8bf*)lds = *(const v8bf*)src;                 // VGPR-bounce fallback
#endif
}

__device__ __forceinline__ void async_wait_all() {
#ifdef HAVE_ASYNC_LDS
#  ifdef HAVE_WAIT_ASYNC
    __builtin_amdgcn_s_wait_asynccnt(0);
#  else
    asm volatile("s_wait_asynccnt 0" ::: "memory");
#  endif
#endif
}

__device__ __forceinline__ unsigned short f32_to_bf16(float f) {
    union { float f; unsigned int u; } c; c.f = f;
    unsigned int u = c.u;
    u += 0x7fffu + ((u >> 16) & 1u);           // round-to-nearest-even
    return (unsigned short)(u >> 16);
}

__device__ __forceinline__ v16bf cat8(v8bf lo, v8bf hi) {
    return __builtin_shufflevector(lo, hi, 0,1,2,3,4,5,6,7,8,9,10,11,12,13,14,15);
}

__device__ __forceinline__ v8f wmma_bf16(v16bf a, v16bf b, v8f c) {
    return __builtin_amdgcn_wmma_f32_16x16x32_bf16(false, a, false, b, (short)0, c,
                                                   false, false);
}

__device__ __forceinline__ float block_sum_256(float v) {
    __shared__ float red[8];
    int lane = threadIdx.x & 31, wv = threadIdx.x >> 5;
    #pragma unroll
    for (int o = 16; o > 0; o >>= 1) v += __shfl_xor(v, o, 32);
    if (lane == 0) red[wv] = v;
    __syncthreads();
    float s = (threadIdx.x < 8) ? red[threadIdx.x] : 0.0f;
    if (wv == 0) {
        #pragma unroll
        for (int o = 4; o > 0; o >>= 1) s += __shfl_xor(s, o, 32);
        if (lane == 0) red[0] = s;
    }
    __syncthreads();
    return red[0];
}

// -------------------------------------------------- weight fp32 -> bf16^T --
__global__ __launch_bounds__(256) void transpose_bf16_k(
    const float* __restrict__ W, unsigned short* __restrict__ Wt, int K, int N)
{
    __shared__ unsigned short tile[32][33];
    int n0 = blockIdx.x * 32, k0 = blockIdx.y * 32;
    int tx = threadIdx.x & 31, ty = threadIdx.x >> 5;
    #pragma unroll
    for (int i = 0; i < 32; i += 8)
        tile[ty + i][tx] = f32_to_bf16(W[(size_t)(k0 + ty + i) * N + n0 + tx]);
    __syncthreads();
    #pragma unroll
    for (int i = 0; i < 32; i += 8)
        Wt[(size_t)(n0 + ty + i) * K + k0 + tx] = tile[tx][ty + i];
}

// ------------------------------------------------------------- rmsnorm -----
__global__ __launch_bounds__(256) void rmsnorm_bf16_k(
    const float* __restrict__ X, const float* __restrict__ w,
    unsigned short* __restrict__ Y)
{
    const float* x = X + (size_t)blockIdx.x * HID;
    unsigned short* y = Y + (size_t)blockIdx.x * HID;
    float ss = 0.0f;
    for (int i = threadIdx.x; i < HID; i += 256) { float v = x[i]; ss += v * v; }
    ss = block_sum_256(ss);
    float sc = rsqrtf(ss * (1.0f / HID) + 1e-6f);
    for (int i = threadIdx.x; i < HID; i += 256)
        y[i] = f32_to_bf16(x[i] * sc * (1.0f + w[i]));
}

__global__ __launch_bounds__(256) void addnorm_f32_k(
    const float* __restrict__ res, const float* __restrict__ T,
    const float* __restrict__ w, float* __restrict__ out)
{
    const float* t = T + (size_t)blockIdx.x * HID;
    const float* r = res + (size_t)blockIdx.x * HID;
    float* o = out + (size_t)blockIdx.x * HID;
    float ss = 0.0f;
    for (int i = threadIdx.x; i < HID; i += 256) { float v = t[i]; ss += v * v; }
    ss = block_sum_256(ss);
    float sc = rsqrtf(ss * (1.0f / HID) + 1e-6f);
    for (int i = threadIdx.x; i < HID; i += 256)
        o[i] = r[i] + t[i] * sc * (1.0f + w[i]);
}

// ------------------------------------------------------ generic WMMA GEMM --
// C[M][N](f32) = A[M][K](bf16 rm) * Bt[N][K](bf16). Tile 128x128, 8 waves.
// Double-buffered LDS B-tile, async global->LDS staging, 1 barrier / k-step,
// last k-step peeled so steady-state staging is branch-free.
// grid = (N/128, M/128)
__global__ __launch_bounds__(256) void gemm_bf16_k(
    const unsigned short* __restrict__ A, const unsigned short* __restrict__ Bt,
    float* __restrict__ C, int M, int N, int K)
{
    __shared__ alignas(32) unsigned short bsh[2][128 * 32];     // 16 KB
    const int tid  = threadIdx.x;
    const int wave = tid >> 5, lane = tid & 31;
    const int l15  = lane & 15, hi = lane >> 4;
    const int rowBase = blockIdx.y * 128 + wave * 16;
    const int colBase = blockIdx.x * 128;
    v8f acc[8] = {};

    const int sr = tid >> 1;                 // 0..127 : staged Bt row
    const int sc = (tid & 1) * 16;           // 0 / 16 : staged k offset
    const unsigned short* bsrc = Bt + (size_t)(colBase + sr) * K + sc;
    const unsigned short* arow = A + (size_t)(rowBase + l15) * K + 16 * hi;

    int k = 0;
    auto compute = [&](const unsigned short* bbase) {
        // A fragment: halfs0-15 K=k+16*hi+0..15 (matches B)
        v8bf alo = *(const v8bf*)(arow + k);
        v8bf ahi = *(const v8bf*)(arow + k + 8);
        v16bf afrag = cat8(alo, ahi);
        v16bf bf[8];
        #pragma unroll
        for (int t = 0; t < 8; t++)          // batch all LDS fragment reads
            bf[t] = *(const v16bf*)&bbase[(t * 16 + l15) * 32 + 16 * hi];
        #pragma unroll
        for (int t = 0; t < 8; t++)
            acc[t] = wmma_bf16(afrag, bf[t], acc[t]);
    };

    // prologue: stage tile k=0 into buffer 0
    stage16(bsrc,     &bsh[0][sr * 32 + sc]);
    stage16(bsrc + 8, &bsh[0][sr * 32 + sc + 8]);

    int c = 0;
    for (; k < K - 32; k += 32) {            // steady state: unconditional stage
        async_wait_all();            // buf[c] resident (own async ops done)
        __syncthreads();             // all waves staged; prior reads retired
        stage16(bsrc + k + 32,     &bsh[c ^ 1][sr * 32 + sc]);
        stage16(bsrc + k + 40,     &bsh[c ^ 1][sr * 32 + sc + 8]);
        __builtin_prefetch(arow + k + 32, 0, 3);           // global_prefetch_b8
        compute(bsh[c]);
        c ^= 1;
    }
    async_wait_all();                        // epilogue: last tile, no staging
    __syncthreads();
    compute(bsh[c]);

    #pragma unroll
    for (int t = 0; t < 8; t++)
        #pragma unroll
        for (int r = 0; r < 8; r++)
            C[(size_t)(rowBase + r + 8 * hi) * N + colBase + t * 16 + l15] = acc[t][r];
}

// ------------------------------------------------- rope / split / layout ---
__global__ __launch_bounds__(256) void rope_split_k(
    const float* __restrict__ qkv, const float* __restrict__ cosT,
    const float* __restrict__ sinT, unsigned short* __restrict__ Qb,
    unsigned short* __restrict__ Kb, unsigned short* __restrict__ Vt)
{
    const int s = blockIdx.x;
    const float* row = qkv + (size_t)s * QKV_N;
    const float* cs = cosT + (size_t)s * 128;
    const float* sn = sinT + (size_t)s * 128;
    const float scale = 0.0625f;                   // 256^-0.5

    for (int idx = threadIdx.x; idx < NHEAD * 128; idx += 256) {
        int h = idx >> 7, i = idx & 127;
        float x1 = row[h * HD + i], x2 = row[h * HD + 128 + i];
        float c = cs[i], sv = sn[i];
        unsigned short* q = Qb + ((size_t)h * S_LEN + s) * HD;
        q[i]       = f32_to_bf16((x1 * c - x2 * sv) * scale);
        q[i + 128] = f32_to_bf16((x1 * sv + x2 * c) * scale);
    }
    for (int idx = threadIdx.x; idx < NKV * 128; idx += 256) {
        int h = idx >> 7, i = idx & 127;
        const float* kp = row + NHEAD * HD + h * HD;
        float x1 = kp[i], x2 = kp[128 + i];
        float c = cs[i], sv = sn[i];
        unsigned short* kd = Kb + ((size_t)h * S_LEN + s) * HD;
        kd[i]       = f32_to_bf16(x1 * c - x2 * sv);
        kd[i + 128] = f32_to_bf16(x1 * sv + x2 * c);
    }
    for (int idx = threadIdx.x; idx < NKV * HD; idx += 256) {
        int h = idx >> 8, d = idx & 255;
        float v = row[(NHEAD + NKV) * HD + h * HD + d];
        Vt[((size_t)h * HD + d) * S_LEN + s] = f32_to_bf16(v);
    }
}

// --------------------------------------------- flash attention (windowed) --
__global__ __launch_bounds__(128) void attn_k(
    const unsigned short* __restrict__ Qb, const unsigned short* __restrict__ Kb,
    const unsigned short* __restrict__ Vt, unsigned short* __restrict__ Aout)
{
    const int head = blockIdx.y;
    const int kvh  = head >> 1;                     // H/KV == 2
    const int wave = threadIdx.x >> 5, lane = threadIdx.x & 31;
    const int l15 = lane & 15, hi = lane >> 4;
    const int m0 = blockIdx.x * 64 + wave * 16;

    __shared__ alignas(32) unsigned short psh_all[4][16 * 32];
    unsigned short* psh = psh_all[wave];

    const unsigned short* Qh = Qb + (size_t)head * S_LEN * HD;
    const unsigned short* Kh = Kb + (size_t)kvh * S_LEN * HD;
    const unsigned short* Vh = Vt + (size_t)kvh * HD * S_LEN;

    v8f o[16] = {};
    float mrow[8], lrow[8];
    #pragma unroll
    for (int r = 0; r < 8; r++) { mrow[r] = NEGF; lrow[r] = 0.0f; }

    int kstart = m0 - (WIN - 16);
    if (kstart < 0) kstart = 0;
    kstart &= ~31;
    const int kend = m0 + 16;

    for (int kc = kstart; kc < kend; kc += 32) {
        // ---- S(16x32) = Q(16x256) * K^T ---------------------------------
        v8f s0 = {}, s1 = {};
        #pragma unroll
        for (int kk = 0; kk < HD; kk += 32) {
            const unsigned short* ap = Qh + (size_t)(m0 + l15) * HD + kk + 16 * hi;
            v16bf a = cat8(*(const v8bf*)ap, *(const v8bf*)(ap + 8));
            v16bf b0 = *(const v16bf*)(Kh + (size_t)(kc + l15) * HD + kk + 16 * hi);
            v16bf b1 = *(const v16bf*)(Kh + (size_t)(kc + 16 + l15) * HD + kk + 16 * hi);
            s0 = wmma_bf16(a, b0, s0);
            s1 = wmma_bf16(a, b1, s1);
        }
        // ---- mask + soft-cap -------------------------------------------
        #pragma unroll
        for (int r = 0; r < 8; r++) {
            int q = m0 + r + 8 * hi;
            int k0c = kc + l15, k1c = kc + 16 + l15;
            float v0 = tanhf(s0[r] * 0.02f) * 50.0f;
            float v1 = tanhf(s1[r] * 0.02f) * 50.0f;
            s0[r] = (k0c <= q && q - k0c <= WIN - 1) ? v0 : NEGF;
            s1[r] = (k1c <= q && q - k1c <= WIN - 1) ? v1 : NEGF;
        }
        // ---- online softmax update -------------------------------------
        #pragma unroll
        for (int r = 0; r < 8; r++) {
            float mx = fmaxf(s0[r], s1[r]);
            #pragma unroll
            for (int d = 8; d > 0; d >>= 1) mx = fmaxf(mx, __shfl_xor(mx, d, 32));
            float mnew = fmaxf(mrow[r], mx);
            float corr = __expf(mrow[r] - mnew);
            float e0 = (s0[r] <= 0.5f * NEGF) ? 0.0f : __expf(s0[r] - mnew);
            float e1 = (s1[r] <= 0.5f * NEGF) ? 0.0f : __expf(s1[r] - mnew);
            float rs = e0 + e1;
            #pragma unroll
            for (int d = 8; d > 0; d >>= 1) rs += __shfl_xor(rs, d, 32);
            lrow[r] = lrow[r] * corr + rs;
            mrow[r] = mnew;
            #pragma unroll
            for (int t = 0; t < 16; t++) o[t][r] *= corr;
            psh[(r + 8 * hi) * 32 + l15]      = f32_to_bf16(e0);   // C->A relayout
            psh[(r + 8 * hi) * 32 + 16 + l15] = f32_to_bf16(e1);
        }
        asm volatile("s_wait_dscnt 0" ::: "memory");  // cross-lane LDS visibility
        // ---- O += P(16x32) * V(32x256) ---------------------------------
        const unsigned short* pp = &psh[l15 * 32 + 16 * hi];
        v16bf pa = cat8(*(const v8bf*)pp, *(const v8bf*)(pp + 8));
        #pragma unroll
        for (int t = 0; t < 16; t++) {
            v16bf vb = *(const v16bf*)(Vh + (size_t)(t * 16 + l15) * S_LEN + kc + 16 * hi);
            o[t] = wmma_bf16(pa, vb, o[t]);
        }
        asm volatile("s_wait_dscnt 0" ::: "memory");  // before psh rewrite
    }
    // ---- normalize + store [s][h*256+d] bf16 ----------------------------
    #pragma unroll
    for (int r = 0; r < 8; r++) {
        float inv = (lrow[r] > 0.0f) ? 1.0f / lrow[r] : 0.0f;
        size_t row = (size_t)(m0 + r + 8 * hi) * (NHEAD * HD) + head * HD;
        #pragma unroll
        for (int t = 0; t < 16; t++)
            Aout[row + t * 16 + l15] = f32_to_bf16(o[t][r] * inv);
    }
}

// ------------------------------------------------- gelu(gate)*up -> bf16 ---
__global__ __launch_bounds__(256) void act_k(
    const float* __restrict__ g, const float* __restrict__ u,
    unsigned short* __restrict__ act)
{
    size_t i = (size_t)blockIdx.x * 256 + threadIdx.x;
    float x = g[i];
    float t = tanhf(0.7978845608f * (x + 0.044715f * x * x * x));
    act[i] = f32_to_bf16(0.5f * x * (1.0f + t) * u[i]);
}

// ==================================================================== host =
extern "C" void kernel_launch(void* const* d_in, const int* in_sizes, int n_in,
                              void* d_out, int out_size, void* d_ws, size_t ws_size,
                              hipStream_t stream)
{
    const float* hidden       = (const float*)d_in[0];
    const float* qkv_w        = (const float*)d_in[1];
    const float* o_w          = (const float*)d_in[2];
    const float* gate_w       = (const float*)d_in[3];
    const float* up_w         = (const float*)d_in[4];
    const float* down_w       = (const float*)d_in[5];
    const float* in_ln        = (const float*)d_in[6];
    const float* post_attn_ln = (const float*)d_in[7];
    const float* pre_ffw_ln   = (const float*)d_in[8];
    const float* post_ffw_ln  = (const float*)d_in[9];
    const float* cosT         = (const float*)d_in[10];
    const float* sinT         = (const float*)d_in[11];
    float* out = (float*)d_out;

    char* cur = (char*)d_ws;
    auto alloc = [&](size_t bytes) -> void* {
        void* p = (void*)cur; cur += (bytes + 255) & ~(size_t)255; return p;
    };
    unsigned short* qkvwT  = (unsigned short*)alloc((size_t)QKV_N * HID * 2);
    unsigned short* owT    = (unsigned short*)alloc((size_t)HID * (NHEAD*HD) * 2);
    unsigned short* gatewT = (unsigned short*)alloc((size_t)FFDIM * HID * 2);
    unsigned short* upwT   = (unsigned short*)alloc((size_t)FFDIM * HID * 2);
    unsigned short* downwT = (unsigned short*)alloc((size_t)HID * FFDIM * 2);
    unsigned short* xnorm  = (unsigned short*)alloc((size_t)S_LEN * HID * 2);
    float*          qkvf   = (float*)alloc((size_t)S_LEN * QKV_N * 4);
    unsigned short* Qb     = (unsigned short*)alloc((size_t)NHEAD * S_LEN * HD * 2);
    unsigned short* Kb     = (unsigned short*)alloc((size_t)NKV * S_LEN * HD * 2);
    unsigned short* Vt     = (unsigned short*)alloc((size_t)NKV * HD * S_LEN * 2);
    unsigned short* attnb  = (unsigned short*)alloc((size_t)S_LEN * NHEAD * HD * 2);
    float*          aproj  = (float*)alloc((size_t)S_LEN * HID * 4);
    float*          hbuf   = (float*)alloc((size_t)S_LEN * HID * 4);
    unsigned short* x2     = (unsigned short*)alloc((size_t)S_LEN * HID * 2);
    float*          gbuf   = (float*)alloc((size_t)S_LEN * FFDIM * 4);
    float*          ubuf   = (float*)alloc((size_t)S_LEN * FFDIM * 4);
    unsigned short* actb   = (unsigned short*)alloc((size_t)S_LEN * FFDIM * 2);
    float*          dbuf   = (float*)alloc((size_t)S_LEN * HID * 4);

    // 1) weights -> bf16 transposed
    transpose_bf16_k<<<dim3(QKV_N/32, HID/32),   256, 0, stream>>>(qkv_w,  qkvwT,  HID, QKV_N);
    transpose_bf16_k<<<dim3(HID/32, (NHEAD*HD)/32),256,0,stream>>>(o_w,    owT,    NHEAD*HD, HID);
    transpose_bf16_k<<<dim3(FFDIM/32, HID/32),   256, 0, stream>>>(gate_w, gatewT, HID, FFDIM);
    transpose_bf16_k<<<dim3(FFDIM/32, HID/32),   256, 0, stream>>>(up_w,   upwT,   HID, FFDIM);
    transpose_bf16_k<<<dim3(HID/32, FFDIM/32),   256, 0, stream>>>(down_w, downwT, FFDIM, HID);

    // 2) input RMSNorm
    rmsnorm_bf16_k<<<S_LEN, 256, 0, stream>>>(hidden, in_ln, xnorm);

    // 3) QKV GEMM
    gemm_bf16_k<<<dim3(QKV_N/128, S_LEN/128), 256, 0, stream>>>(xnorm, qkvwT, qkvf,
                                                                S_LEN, QKV_N, HID);
    // 4) RoPE + split + V transpose
    rope_split_k<<<S_LEN, 256, 0, stream>>>(qkvf, cosT, sinT, Qb, Kb, Vt);

    // 5) windowed flash attention
    attn_k<<<dim3(S_LEN/64, NHEAD), 128, 0, stream>>>(Qb, Kb, Vt, attnb);

    // 6) output projection
    gemm_bf16_k<<<dim3(HID/128, S_LEN/128), 256, 0, stream>>>(attnb, owT, aproj,
                                                              S_LEN, HID, NHEAD*HD);
    // 7) h = hidden + rmsnorm(aproj)
    addnorm_f32_k<<<S_LEN, 256, 0, stream>>>(hidden, aproj, post_attn_ln, hbuf);
    // 8) x2 = rmsnorm(h)
    rmsnorm_bf16_k<<<S_LEN, 256, 0, stream>>>(hbuf, pre_ffw_ln, x2);

    // 9) FFN
    gemm_bf16_k<<<dim3(FFDIM/128, S_LEN/128), 256, 0, stream>>>(x2, gatewT, gbuf,
                                                                S_LEN, FFDIM, HID);
    gemm_bf16_k<<<dim3(FFDIM/128, S_LEN/128), 256, 0, stream>>>(x2, upwT, ubuf,
                                                                S_LEN, FFDIM, HID);
    act_k<<<(S_LEN * (size_t)FFDIM) / 256, 256, 0, stream>>>(gbuf, ubuf, actb);
    gemm_bf16_k<<<dim3(HID/128, S_LEN/128), 256, 0, stream>>>(actb, downwT, dbuf,
                                                              S_LEN, HID, FFDIM);
    // 10) out = h + rmsnorm(ffn)
    addnorm_f32_k<<<S_LEN, 256, 0, stream>>>(hbuf, dbuf, post_ffw_ln, out);
}
